// PatientMILFeatures_62904091018117
// MI455X (gfx1250) — matrix operand order repaired and checked by reading, not compiled, not run on hardware
//
#include <hip/hip_runtime.h>
#include <hip/hip_bf16.h>
#include <math.h>

// Problem constants (match reference)
#define B_  32
#define N_  8192
#define D_  1024
#define H_  128

// WMMA fragment types (CDNA5, wave32)
typedef __attribute__((ext_vector_type(16))) __bf16 v16bf;
typedef __attribute__((ext_vector_type(8)))  __bf16 v8bf;
typedef __attribute__((ext_vector_type(8)))  float  v8f;

__device__ __forceinline__ float fast_tanh(float x) {
#if __has_builtin(__builtin_amdgcn_tanhf)
    return __builtin_amdgcn_tanhf(x);     // gfx1250 hardware TRANS tanh
#else
    return tanhf(x);
#endif
}

// -------- Kernel 1: logits_attn[b,n] = mask ? (tanh(feats@w1+b1)@w2 + b2) : -1e9
// Grid: (N/128, B), 256 threads (8 waves); each wave computes a 16-row tile of
// h = feats@w1 with v_wmma_f32_16x16x32_bf16, epilogue fused in registers.
// Software pipelining: A raw-fp32 prefetched one k-step ahead (global latency
// covered by the 8-WMMA chain); B fragments double-buffered two column tiles
// ahead (LDS latency covered by WMMA execution).
#define KSLAB 128
#define PITCH (KSLAB + 8)   // bf16 elems per LDS row; +8 keeps 16B align & spreads banks

__global__ __launch_bounds__(256)
void attn_logits_kernel(const float* __restrict__ feats,
                        const unsigned char* __restrict__ mask,
                        const float* __restrict__ w1,
                        const float* __restrict__ b1,
                        const float* __restrict__ w2,
                        const float* __restrict__ b2,
                        float* __restrict__ logitsA)
{
    __shared__ __bf16 w1s[H_ * PITCH];          // 128 cols x 136 = ~34 KB

    const int tid     = threadIdx.x;
    const int lane    = tid & 31;
    const int wave    = tid >> 5;
    const int b       = blockIdx.y;
    const int rowBase = blockIdx.x * 128 + wave * 16;   // 16 feats rows per wave
    const int halfSel = lane >> 4;                      // 0: lanes 0-15, 1: lanes 16-31
    const int lane16  = lane & 15;

    // A-matrix row for this lane (lanes L and L+16 hold the same row, different K chunks)
    const size_t featRow = ((size_t)b * N_ + (rowBase + lane16)) * (size_t)D_;

    v8f acc[8] = {};   // 8 column tiles of 16 -> H=128; f32 accumulators

    for (int k0 = 0; k0 < D_; k0 += KSLAB) {
        // Cooperative load of w1[k0:k0+KSLAB, 0:128] into LDS, transposed, as bf16.
        // Global reads coalesced along H; LDS writes strided (once per slab).
        for (int i = tid; i < KSLAB * H_; i += 256) {
            const int kk = i >> 7;      // / 128
            const int c  = i & 127;
            w1s[c * PITCH + kk] = (__bf16)w1[(size_t)(k0 + kk) * H_ + c];
        }
        __syncthreads();

        // Preload raw A data for the first k-step of this slab.
        v8f alo = *(const v8f*)(feats + featRow + k0 + halfSel * 8);
        v8f ahi = *(const v8f*)(feats + featRow + k0 + halfSel * 8 + 16);

        #pragma unroll
        for (int ks = 0; ks < KSLAB; ks += 32) {
            // ---- A fragment (16x32 bf16): per ISA layout lanes 0-15 hold K
            // [+0,+8) and [+16,+24); lanes 16-31 hold [+8,+16) and [+24,+32).
            v16bf afrag;
            #pragma unroll
            for (int i = 0; i < 8; ++i) {
                afrag[i]     = (__bf16)alo[i];
                afrag[i + 8] = (__bf16)ahi[i];
            }
            // Prefetch next k-step's raw A while the WMMA chain runs.
            if (ks + 32 < KSLAB) {
                const int cA = k0 + ks + 32 + halfSel * 8;
                alo = *(const v8f*)(feats + featRow + cA);
                ahi = *(const v8f*)(feats + featRow + cA + 16);
            }

            // ---- B fragments (32x16 bf16) from transposed LDS, pipelined 2 deep.
            // Lane = column; lanes 0-15 K [0,8)+[16,24), lanes 16-31 K [8,16)+[24,32).
            const __bf16* brow = &w1s[lane16 * PITCH + ks + halfSel * 8];
            v8bf bl0 = *(const v8bf*)(brow);
            v8bf bh0 = *(const v8bf*)(brow + 16);
            v8bf bl1 = *(const v8bf*)(brow + 16 * PITCH);
            v8bf bh1 = *(const v8bf*)(brow + 16 * PITCH + 16);

            #pragma unroll
            for (int ct = 0; ct < 8; ++ct) {
                v16bf bfrag;
                #pragma unroll
                for (int i = 0; i < 8; ++i) {
                    bfrag[i]     = bl0[i];
                    bfrag[i + 8] = bh0[i];
                }
                bl0 = bl1; bh0 = bh1;
                if (ct + 2 < 8) {
                    const __bf16* bp = brow + (ct + 2) * 16 * PITCH;
                    bl1 = *(const v8bf*)bp;
                    bh1 = *(const v8bf*)(bp + 16);
                }
                acc[ct] = __builtin_amdgcn_wmma_f32_16x16x32_bf16(
                    /*neg_a=*/false, afrag, /*neg_b=*/false, bfrag,
                    /*c_mod=*/(short)0, acc[ct], /*reuse_a=*/false, /*reuse_b=*/false);
            }
        }
        __syncthreads();
    }

    // ---- Fused epilogue: part[v] = sum_h tanh(h[row,h] + b1[h]) * w2[h]
    // C/D layout: VGPR v <-> row v + 8*halfSel, column = lane16 (+16*ct).
    float part[8];
    #pragma unroll
    for (int v = 0; v < 8; ++v) part[v] = 0.f;

    #pragma unroll
    for (int ct = 0; ct < 8; ++ct) {
        const int col   = ct * 16 + lane16;
        const float w2v = w2[col];
        const float b1v = b1[col];
        #pragma unroll
        for (int v = 0; v < 8; ++v)
            part[v] += fast_tanh(acc[ct][v] + b1v) * w2v;
    }

    // Reduce across the 16 lanes holding the same rows (two independent halves).
    #pragma unroll
    for (int v = 0; v < 8; ++v) {
        part[v] += __shfl_xor(part[v], 1, 16);
        part[v] += __shfl_xor(part[v], 2, 16);
        part[v] += __shfl_xor(part[v], 4, 16);
        part[v] += __shfl_xor(part[v], 8, 16);
    }

    if (lane16 == 0) {
        const float b2v = b2[0];
        #pragma unroll
        for (int v = 0; v < 8; ++v) {
            const int n = rowBase + halfSel * 8 + v;
            const float lg = part[v] + b2v;
            const size_t idx = (size_t)b * N_ + n;
            logitsA[idx] = mask[idx] ? lg : -1e9f;
        }
    }
}

// -------- Kernel 2: masked softmax over N per batch + mask-renormalization.
__global__ __launch_bounds__(256)
void softmax_kernel(const float* __restrict__ logitsA,
                    const unsigned char* __restrict__ mask,
                    float* __restrict__ attn)
{
    __shared__ float red[256];
    const int b = blockIdx.x, tid = threadIdx.x;
    const float* x = logitsA + (size_t)b * N_;
    const unsigned char* m = mask + (size_t)b * N_;
    float* a = attn + (size_t)b * N_;

    float mx = -3.4e38f;
    for (int n = tid; n < N_; n += 256) mx = fmaxf(mx, x[n]);
    red[tid] = mx; __syncthreads();
    for (int s = 128; s > 0; s >>= 1) { if (tid < s) red[tid] = fmaxf(red[tid], red[tid + s]); __syncthreads(); }
    mx = red[0]; __syncthreads();

    float sum = 0.f;
    for (int n = tid; n < N_; n += 256) sum += __expf(x[n] - mx);
    red[tid] = sum; __syncthreads();
    for (int s = 128; s > 0; s >>= 1) { if (tid < s) red[tid] += red[tid + s]; __syncthreads(); }
    const float inv = 1.f / red[0]; __syncthreads();

    float s2 = 0.f;
    for (int n = tid; n < N_; n += 256) {
        float p = __expf(x[n] - mx) * inv;
        p = m[n] ? p : 0.f;
        a[n] = p;
        s2 += p;
    }
    red[tid] = s2; __syncthreads();
    for (int s = 128; s > 0; s >>= 1) { if (tid < s) red[tid] += red[tid + s]; __syncthreads(); }
    const float scale = 1.f / fmaxf(red[0], 1e-8f);
    for (int n = tid; n < N_; n += 256) a[n] *= scale;
}

// -------- Kernel 3: pooled[b,d] = sum_n attn[b,n] * feats[b,n,d]
// Streaming, coalesced across d; second (and last) full read of feats.
__global__ __launch_bounds__(256)
void pool_kernel(const float* __restrict__ feats,
                 const float* __restrict__ attn,
                 float* __restrict__ pooled)
{
    const int b = blockIdx.y;
    const int d = blockIdx.x * 256 + threadIdx.x;
    const float* fb = feats + (size_t)b * N_ * D_ + d;
    const float* ab = attn + (size_t)b * N_;
    float acc = 0.f;
    for (int n = 0; n < N_; n += 4) {
        acc += ab[n + 0] * fb[(size_t)(n + 0) * D_];
        acc += ab[n + 1] * fb[(size_t)(n + 1) * D_];
        acc += ab[n + 2] * fb[(size_t)(n + 2) * D_];
        acc += ab[n + 3] * fb[(size_t)(n + 3) * D_];
    }
    pooled[(size_t)b * D_ + d] = acc;
}

// -------- Kernel 4: logits[b] = pooled[b,:] @ wc + bc
__global__ __launch_bounds__(256)
void classify_kernel(const float* __restrict__ pooled,
                     const float* __restrict__ wc,
                     const float* __restrict__ bc,
                     float* __restrict__ logits)
{
    __shared__ float red[256];
    const int b = blockIdx.x, tid = threadIdx.x;
    float s = 0.f;
    for (int d = tid; d < D_; d += 256) s += pooled[(size_t)b * D_ + d] * wc[d];
    red[tid] = s; __syncthreads();
    for (int st = 128; st > 0; st >>= 1) { if (tid < st) red[tid] += red[tid + st]; __syncthreads(); }
    if (tid == 0) logits[b] = red[0] + bc[0];
}

extern "C" void kernel_launch(void* const* d_in, const int* in_sizes, int n_in,
                              void* d_out, int out_size, void* d_ws, size_t ws_size,
                              hipStream_t stream)
{
    const float*         feats = (const float*)d_in[0];
    const unsigned char* mask  = (const unsigned char*)d_in[1];  // jnp.bool_ = 1 byte
    const float*         w1    = (const float*)d_in[2];
    const float*         b1    = (const float*)d_in[3];
    const float*         w2    = (const float*)d_in[4];
    const float*         b2    = (const float*)d_in[5];
    const float*         wc    = (const float*)d_in[6];
    const float*         bc    = (const float*)d_in[7];

    float* out     = (float*)d_out;
    float* logits  = out;                    // [B, 1]
    float* pooled  = out + B_;               // [B, D]
    float* attn    = out + B_ + B_ * D_;     // [B, N]
    float* logitsA = (float*)d_ws;           // [B, N] masked attention logits

    dim3 g1(N_ / 128, B_);
    attn_logits_kernel<<<g1, 256, 0, stream>>>(feats, mask, w1, b1, w2, b2, logitsA);

    softmax_kernel<<<B_, 256, 0, stream>>>(logitsA, mask, attn);

    dim3 g3(D_ / 256, B_);
    pool_kernel<<<g3, 256, 0, stream>>>(feats, attn, pooled);

    classify_kernel<<<B_, 256, 0, stream>>>(pooled, wc, bc, logits);
}
